// PointScoringHead_28759101014142
// MI455X (gfx1250) — matrix operand order repaired and checked by reading, not compiled, not run on hardware
//
#include <hip/hip_runtime.h>

typedef __attribute__((ext_vector_type(2)))  float  v2f;
typedef __attribute__((ext_vector_type(8)))  float  v8f;
typedef __attribute__((ext_vector_type(8)))  __bf16 v8bf;
typedef __attribute__((ext_vector_type(16))) __bf16 v16bf;

#define FEAT   256
#define GDIM   259          // FEAT + 3
#define DIN    516          // 259 + 1 + 256
#define HID    128
#define KCENT  2048
#define KSTEPS 17           // ceil(516/32)
#define ROWS   128          // rows per block (8 waves x 16)
#define STRIDE 552          // padded LDS row stride (bf16 elems), conflict-free
#define NTILES 8            // 128 / 16

// Low 32 bits of a flat shared-aperture address == raw LDS byte offset
// (ISA 10.2).
__device__ __forceinline__ unsigned lds_off(const void* p) {
  return (unsigned)(size_t)p;
}

// CDNA5 async memory->LDS copy (ASYNCcnt-tracked, no VGPR round trip).
__device__ __forceinline__ void async_copy_b128(unsigned ldsByteOff,
                                                unsigned long long gaddr) {
  asm volatile("global_load_async_to_lds_b128 %0, %1, off"
               :: "v"(ldsByteOff), "v"(gaddr) : "memory");
}
__device__ __forceinline__ void async_copy_b32(unsigned ldsByteOff,
                                               unsigned long long gaddr) {
  asm volatile("global_load_async_to_lds_b32 %0, %1, off"
               :: "v"(ldsByteOff), "v"(gaddr) : "memory");
}
__device__ __forceinline__ void wait_async0() {
  asm volatile("s_wait_asynccnt 0x0" ::: "memory");
}

// ---------------------------------------------------------------------------
// Kernel 1: build WMMA-ready operand layouts.
//  Bmat[ct][lane][v]  : f32 B-fragment for V_WMMA_F32_16X16X4_F32.
//                       col n = ct*16 + lane%16, K = 2*(lane/16)+v,
//                       rows 0..2 = -mu, row 3 = 0.5*||mu||^2  (score = 0.5||mu||^2 - p.mu)
//  w1f[kk][nt][lane][s]: bf16 B-fragment for V_WMMA_F32_16X16X32_BF16.
//                       col N = nt*16 + lane%16, K = kk*32 + (lane/16)*16 + s, zero-pad K>=516.
// ---------------------------------------------------------------------------
__global__ void prep_kernel(const float* __restrict__ mu, const float* __restrict__ W1,
                            float* __restrict__ Bmat, __bf16* __restrict__ w1f) {
  int tid = blockIdx.x * blockDim.x + threadIdx.x;
  int stride = gridDim.x * blockDim.x;

  for (int e = tid; e < (KCENT / 16) * 32 * 2; e += stride) {
    int v = e & 1, lane = (e >> 1) & 31, ct = e >> 6;
    int n = lane & 15, gg = lane >> 4;
    int c = ct * 16 + n;
    int kk = 2 * gg + v;
    float val;
    if (kk < 3) {
      val = -mu[c * 3 + kk];
    } else {
      float x = mu[c * 3], y = mu[c * 3 + 1], z = mu[c * 3 + 2];
      val = 0.5f * (x * x + y * y + z * z);
    }
    Bmat[e] = val;
  }

  for (int e = tid; e < KSTEPS * NTILES * 32 * 16; e += stride) {
    int s = e & 15, lane = (e >> 4) & 31, nt = (e >> 9) & 7, kk = e >> 12;
    int K = kk * 32 + (lane >> 4) * 16 + s;
    int Ncol = nt * 16 + (lane & 15);
    float val = (K < DIN) ? W1[K * HID + Ncol] : 0.0f;
    w1f[e] = (__bf16)val;
  }
}

// ---------------------------------------------------------------------------
// Kernel 2: 1-NN via V_WMMA_F32_16X16X4_F32.
// Each wave owns 16 points (A fragment resident), loops 128 centroid tiles.
// Argmin tracked with tile-id embedded in score mantissa low 7 bits + v_min.
// B tiles staged global->LDS with CDNA5 async copies.
// ---------------------------------------------------------------------------
__global__ void nn_kernel(const float* __restrict__ p, const float* __restrict__ Bmat,
                          int* __restrict__ idxOut, int N) {
  __shared__ float sB[(KCENT / 16) * 64];   // 32 KB
  int tid = threadIdx.x;
  int wave = tid >> 5, lane = tid & 31;
  int g16 = lane >> 4, n16 = lane & 15;
  int rowBase = blockIdx.x * ROWS + wave * 16;
  int row = rowBase + n16;

  // A fragment first (independent of LDS): lane holds A[M=lane%16][K=2*(lane/16)+v]
  int c0 = 2 * g16, c1 = 2 * g16 + 1;
  v2f a;
  a.x = (row < N) ? p[(size_t)row * 3 + c0] : 0.0f;
  a.y = (c1 < 3) ? ((row < N) ? p[(size_t)row * 3 + c1] : 0.0f) : 1.0f;

  // Stage Bmat (32 KB) into LDS: 2048 x b128 async copies, 8 per thread.
  {
    unsigned base = lds_off(sB);
    unsigned long long gb = (unsigned long long)(size_t)Bmat;
    for (int i = tid; i < (KCENT / 16) * 16; i += blockDim.x)
      async_copy_b128(base + (unsigned)i * 16u, gb + (unsigned long long)i * 16ull);
    wait_async0();
  }
  __syncthreads();

  float best[8];
#pragma unroll
  for (int r = 0; r < 8; ++r) best[r] = __uint_as_float(0x7F000000u);  // huge finite

  for (int ct = 0; ct < KCENT / 16; ++ct) {
    v2f b = *(const v2f*)(sB + ct * 64 + lane * 2);
    v8f s = {};
    s = __builtin_amdgcn_wmma_f32_16x16x4_f32(false, a, false, b, (short)0, s,
                                              false, false);
#pragma unroll
    for (int r = 0; r < 8; ++r) {
      unsigned key = (__float_as_uint(s[r]) & 0xFFFFFF80u) | (unsigned)ct;
      best[r] = fminf(best[r], __uint_as_float(key));
    }
  }

  // cross-lane argmin over the 16 column-lanes of each half-wave
  const int offs[4] = {1, 2, 4, 8};
#pragma unroll
  for (int r = 0; r < 8; ++r) {
    float val = best[r];
    int cidx = (int)((__float_as_uint(val) & 127u) * 16u) + n16;
#pragma unroll
    for (int o = 0; o < 4; ++o) {
      float ov = __shfl_xor(val, offs[o], 32);
      int oc = __shfl_xor(cidx, offs[o], 32);
      if (ov < val) { val = ov; cidx = oc; }
    }
    if (n16 == 0) {
      int rg = rowBase + g16 * 8 + r;   // C/D layout: VGPR r -> row 8*(lane/16)+r
      if (rg < N) idxOut[rg] = cidx;
    }
  }
}

// ---------------------------------------------------------------------------
// Kernel 3: fused gather + MLP via V_WMMA_F32_16X16X32_BF16.
// idx tile fetched with async LDS copy; input tile staged as bf16 in LDS;
// GEMM vs pre-swizzled W1 frags; fused bias + ReLU + W2 dot + reduction.
// ---------------------------------------------------------------------------
__global__ void mlp_kernel(const float* __restrict__ g, const float* __restrict__ rho,
                           const float* __restrict__ h, const int* __restrict__ idx,
                           const __bf16* __restrict__ w1f, const float* __restrict__ b1,
                           const float* __restrict__ W2, const float* __restrict__ b2,
                           float* __restrict__ out, int N) {
  extern __shared__ char smem[];
  int* sIdx = (int*)smem;
  __bf16* sInp = (__bf16*)(smem + 512);

  int tid = threadIdx.x;
  int base = blockIdx.x * ROWS;

  // Async-fetch the 128 gather indices into LDS. Clamp the address for padding
  // rows (their idx is never dereferenced: all reads of sIdx are under rg < N).
  if (tid < ROWS) {
    int rg = base + tid;
    int rc = (rg < N) ? rg : (N - 1);
    async_copy_b32(lds_off(sIdx) + (unsigned)tid * 4u,
                   (unsigned long long)(size_t)(idx + rc));
  }
  wait_async0();
  __syncthreads();

  // stage input tile: [g(259) | rho(1) | h(256) | pad]
  for (int e = tid; e < ROWS * STRIDE; e += blockDim.x) {
    int row = e / STRIDE;
    int f = e - row * STRIDE;
    int rg = base + row;
    float val = 0.0f;
    if (rg < N) {
      if (f < GDIM)       val = g[(size_t)rg * GDIM + f];
      else if (f == GDIM) val = rho[sIdx[row]];
      else if (f < DIN)   val = h[(size_t)sIdx[row] * FEAT + (f - GDIM - 1)];
    }
    sInp[e] = (__bf16)val;
  }
  __syncthreads();

  int wave = tid >> 5, lane = tid & 31;
  int g16 = lane >> 4, n16 = lane & 15;
  const __bf16* rowPtr = sInp + (size_t)(wave * 16 + n16) * STRIDE;

  v8f acc[NTILES] = {};
  for (int kk = 0; kk < KSTEPS; ++kk) {
    int k0 = kk * 32;
    if (kk + 1 < KSTEPS)   // pull next k-step's W1 fragments toward L2/L0
      __builtin_prefetch(w1f + ((size_t)(kk + 1) * NTILES * 32 + lane) * 16, 0, 0);
    // A fragment (contiguous k order, same as w1f): lane group g holds K = g*16+{0..15}
    v8bf alo = *(const v8bf*)(rowPtr + k0 + g16 * 16);
    v8bf ahi = *(const v8bf*)(rowPtr + k0 + g16 * 16 + 8);
    v16bf afrag = __builtin_shufflevector(alo, ahi, 0, 1, 2, 3, 4, 5, 6, 7, 8, 9,
                                          10, 11, 12, 13, 14, 15);
#pragma unroll
    for (int nt = 0; nt < NTILES; ++nt) {
      const v8bf* bp =
          (const v8bf*)(w1f + (((size_t)kk * NTILES + nt) * 32 + lane) * 16);
      v8bf blo = bp[0];
      v8bf bhi = bp[1];
      v16bf bfrag = __builtin_shufflevector(blo, bhi, 0, 1, 2, 3, 4, 5, 6, 7, 8, 9,
                                            10, 11, 12, 13, 14, 15);
      acc[nt] = __builtin_amdgcn_wmma_f32_16x16x32_bf16(false, afrag, false, bfrag,
                                                        (short)0, acc[nt], false,
                                                        false);
    }
  }

  // layer 2: bias + ReLU + dot with W2, accumulated per (row, lane-column-class)
  float partial[8] = {0.f, 0.f, 0.f, 0.f, 0.f, 0.f, 0.f, 0.f};
#pragma unroll
  for (int nt = 0; nt < NTILES; ++nt) {
    float bias = b1[nt * 16 + n16];
    float w2v = W2[nt * 16 + n16];
#pragma unroll
    for (int r = 0; r < 8; ++r) {
      float hv = fmaxf(acc[nt][r] + bias, 0.0f);
      partial[r] = fmaf(hv, w2v, partial[r]);
    }
  }

  const int offs[4] = {1, 2, 4, 8};
#pragma unroll
  for (int r = 0; r < 8; ++r) {
#pragma unroll
    for (int o = 0; o < 4; ++o) partial[r] += __shfl_xor(partial[r], offs[o], 32);
  }

  if (n16 == 0) {
    float bb = b2[0];
#pragma unroll
    for (int r = 0; r < 8; ++r) {
      int rg = base + wave * 16 + g16 * 8 + r;
      if (rg < N) out[rg] = partial[r] + bb;
    }
  }
}

// ---------------------------------------------------------------------------
extern "C" void kernel_launch(void* const* d_in, const int* in_sizes, int n_in,
                              void* d_out, int out_size, void* d_ws, size_t ws_size,
                              hipStream_t stream) {
  const float* p   = (const float*)d_in[0];
  const float* mu  = (const float*)d_in[1];
  const float* rho = (const float*)d_in[2];
  const float* h   = (const float*)d_in[3];
  const float* g   = (const float*)d_in[4];
  const float* W1  = (const float*)d_in[5];
  const float* b1  = (const float*)d_in[6];
  const float* W2  = (const float*)d_in[7];
  const float* b2  = (const float*)d_in[8];
  float* out = (float*)d_out;

  int N = in_sizes[0] / 3;

  char* ws = (char*)d_ws;
  size_t idxBytes = ((size_t)N * sizeof(int) + 255) & ~(size_t)255;
  int* idx = (int*)ws;
  float* Bmat = (float*)(ws + idxBytes);
  size_t bmatBytes = (size_t)(KCENT / 16) * 64 * sizeof(float);  // 32 KB
  __bf16* w1f = (__bf16*)(ws + idxBytes + bmatBytes);

  int nb = (N + ROWS - 1) / ROWS;

  prep_kernel<<<64, 256, 0, stream>>>(mu, W1, Bmat, w1f);
  nn_kernel<<<nb, 256, 0, stream>>>(p, Bmat, idx, N);
  size_t smemBytes = 512 + (size_t)ROWS * STRIDE * sizeof(__bf16);  // ~142 KB
  mlp_kernel<<<nb, 256, smemBytes, stream>>>(g, rho, h, idx, w1f, b1, W2, b2, out, N);
}
